// AttentionPooling_51041391345714
// MI455X (gfx1250) — compile-verified, matrix operand order checked by reading, not run on hardware
//
#include <hip/hip_runtime.h>
#include <hip/hip_bf16.h>

// ---------------------------------------------------------------------------
// Problem constants (fixed by setup_inputs())
// ---------------------------------------------------------------------------
#define B_     4
#define S_     512
#define H_     256
#define NH_    4
#define DH_    64
#define WSPAN_ 16
#define NSPAN_ (S_ * WSPAN_)   // 8192 spans per batch
#define MROWS_ (B_ * NSPAN_)   // 32768 total rows
#define FF_    (4 * H_)        // 1024

typedef __attribute__((ext_vector_type(16))) __bf16 v16bf;
typedef __attribute__((ext_vector_type(8)))  float  v8f;

// ---------------------------------------------------------------------------
// helpers
// ---------------------------------------------------------------------------
static __device__ __forceinline__ unsigned short f2bf(float f) {
  unsigned int u = __float_as_uint(f);
  u += 0x7FFFu + ((u >> 16) & 1u);       // round-to-nearest-even
  return (unsigned short)(u >> 16);
}

// Load one bf16 WMMA fragment (A-style or B^T-style: "row held per lane").
// 16-bit 16x32 layout: lane L<16 -> row M=L, K = {k0..k0+7, k0+16..k0+23};
// lanes 16..31 -> row M=L-16, K shifted by +8.  Two contiguous 16B runs.
static __device__ __forceinline__ v16bf load_frag(const unsigned short* __restrict__ rowp,
                                                  int k0, int lane) {
  union { v16bf v; uint4 q[2]; } f;
  const unsigned short* p = rowp + k0 + ((lane >= 16) ? 8 : 0);
  f.q[0] = *reinterpret_cast<const uint4*>(p);
  f.q[1] = *reinterpret_cast<const uint4*>(p + 16);
  return f.v;
}

// ---------------------------------------------------------------------------
// f32 -> bf16 conversion (weights)
// ---------------------------------------------------------------------------
__global__ void cvt_bf16_kernel(const float* __restrict__ src,
                                unsigned short* __restrict__ dst, int n) {
  int i = blockIdx.x * 256 + threadIdx.x;
  if (i < n) dst[i] = f2bf(src[i]);
}

// ---------------------------------------------------------------------------
// x = token_reps + positional encoding, stored bf16 (GEMM A operand)
// ---------------------------------------------------------------------------
__global__ void prep_kernel(const float* __restrict__ tok,
                            unsigned short* __restrict__ xb) {
  int i = blockIdx.x * 256 + threadIdx.x;          // < B*S*H
  int f = i & (H_ - 1);
  int s = (i >> 8) & (S_ - 1);
  // div = exp(-(2*(f/2)) * ln(10000)/H); pe = even? sin : cos
  float div = expf((float)(f & ~1) * (-9.210340371976184f / (float)H_));
  float arg = (float)s * div;
  float pe  = (f & 1) ? cosf(arg) : sinf(arg);
  xb[i] = f2bf(tok[i] + pe);
}

// ---------------------------------------------------------------------------
// q = dummy_query @ Wq^T + bq   (exact f32, one block)
// ---------------------------------------------------------------------------
__global__ void qproj_kernel(const float* __restrict__ dummy,
                             const float* __restrict__ Wq,
                             const float* __restrict__ bq,
                             float* __restrict__ q) {
  int e = threadIdx.x;                              // 0..255
  const float* wr = Wq + (size_t)e * H_;
  float acc = bq[e];
  for (int h = 0; h < H_; ++h) acc += dummy[h] * wr[h];
  q[e] = acc;
}

// ---------------------------------------------------------------------------
// scores[b,head,s] = dot(q[head], k[b,s,head]) / sqrt(dh)
// ---------------------------------------------------------------------------
__global__ void scores_kernel(const float* __restrict__ q,
                              const float* __restrict__ K,   // (B*S, H) f32
                              float* __restrict__ scores) {  // (B,NH,S)
  int i = blockIdx.x * 256 + threadIdx.x;
  if (i >= B_ * NH_ * S_) return;
  int s    = i % S_;
  int head = (i / S_) % NH_;
  int b    = i / (S_ * NH_);
  const float* kr = K + ((size_t)(b * S_ + s)) * H_ + head * DH_;
  const float* qr = q + head * DH_;
  float acc = 0.f;
  for (int d = 0; d < DH_; ++d) acc += qr[d] * kr[d];
  scores[i] = acc * 0.125f;                         // 1/sqrt(64)
}

// ---------------------------------------------------------------------------
// Per-span masked softmax over <=16 keys + weighted sum of V.
// One block per span, one thread per output channel h (H_=256).
// Output written bf16 (feeds the Wo WMMA GEMM).
// ---------------------------------------------------------------------------
__global__ void span_attn_kernel(const float* __restrict__ scores,   // (B,NH,S)
                                 const float* __restrict__ V,        // (B*S, H)
                                 const int*   __restrict__ span_ids, // (B*N, 2)
                                 unsigned short* __restrict__ ctx_bf) {
  int span = blockIdx.x;                  // 0 .. B*N-1
  int b    = span / NSPAN_;
  int tid  = threadIdx.x;                 // channel h
  int head = tid >> 6;                    // h / dh
  int st = span_ids[span * 2 + 0];
  int en = span_ids[span * 2 + 1];
  const float* sc = scores + ((size_t)b * NH_ + head) * S_;
  float mx = -3.4e38f;
  for (int s = st; s < en; ++s) mx = fmaxf(mx, sc[s]);
  float sum = 0.f, acc = 0.f;
  for (int s = st; s < en; ++s) {
    float w = __expf(sc[s] - mx);
    sum += w;
    acc += w * V[((size_t)(b * S_ + s)) * H_ + tid];
  }
  ctx_bf[(size_t)span * H_ + tid] = f2bf(acc / sum);
}

// ---------------------------------------------------------------------------
// WMMA GEMM:  C[m,n] = sum_k A_bf16[m,k] * W_bf16[n,k] + bias[n]
// 8 waves / block stacked along M.  Each wave owns a 32(M) x 64(N) strip:
// 2 A fragments x 4 B fragments -> 8 v_wmma_f32_16x16x32_bf16 accumulators.
// Per K-step: 12 b128 loads for 8 WMMAs (1.5 VMEM/WMMA issue ratio); each B
// fragment is reused twice from registers.
// ---------------------------------------------------------------------------
template <bool RELU, bool WF32, bool WBF16>
__global__ void gemm_wmma_kernel(const unsigned short* __restrict__ A,
                                 const unsigned short* __restrict__ W,
                                 const float* __restrict__ bias,
                                 float* __restrict__ Cf,
                                 unsigned short* __restrict__ Cb,
                                 int M, int N, int K) {
  const int lane = threadIdx.x & 31;
  const int wave = threadIdx.x >> 5;
  const int m0 = (blockIdx.x * 8 + wave) * 32;     // 32 rows per wave
  const int n0 = blockIdx.y * 64;
  (void)M;

  v8f acc00 = {}, acc01 = {}, acc02 = {}, acc03 = {};
  v8f acc10 = {}, acc11 = {}, acc12 = {}, acc13 = {};

  const unsigned short* arow0 = A + (size_t)(m0 + (lane & 15)) * K;
  const unsigned short* arow1 = arow0 + (size_t)16 * K;
  const unsigned short* wrow0 = W + (size_t)(n0 + (lane & 15)) * K;
  const unsigned short* wrow1 = wrow0 + (size_t)16 * K;
  const unsigned short* wrow2 = wrow0 + (size_t)32 * K;
  const unsigned short* wrow3 = wrow0 + (size_t)48 * K;

  for (int k0 = 0; k0 < K; k0 += 32) {
    v16bf a0 = load_frag(arow0, k0, lane);
    v16bf a1 = load_frag(arow1, k0, lane);
    v16bf b0 = load_frag(wrow0, k0, lane);
    v16bf b1 = load_frag(wrow1, k0, lane);
    v16bf b2 = load_frag(wrow2, k0, lane);
    v16bf b3 = load_frag(wrow3, k0, lane);
    acc00 = __builtin_amdgcn_wmma_f32_16x16x32_bf16(false, a0, false, b0, (short)0, acc00, false, false);
    acc01 = __builtin_amdgcn_wmma_f32_16x16x32_bf16(false, a0, false, b1, (short)0, acc01, false, false);
    acc02 = __builtin_amdgcn_wmma_f32_16x16x32_bf16(false, a0, false, b2, (short)0, acc02, false, false);
    acc03 = __builtin_amdgcn_wmma_f32_16x16x32_bf16(false, a0, false, b3, (short)0, acc03, false, false);
    acc10 = __builtin_amdgcn_wmma_f32_16x16x32_bf16(false, a1, false, b0, (short)0, acc10, false, false);
    acc11 = __builtin_amdgcn_wmma_f32_16x16x32_bf16(false, a1, false, b1, (short)0, acc11, false, false);
    acc12 = __builtin_amdgcn_wmma_f32_16x16x32_bf16(false, a1, false, b2, (short)0, acc12, false, false);
    acc13 = __builtin_amdgcn_wmma_f32_16x16x32_bf16(false, a1, false, b3, (short)0, acc13, false, false);
  }

  // C/D layout: lane L -> col N = L&15; VGPR r -> row M = r + (L>=16 ? 8 : 0)
  const int rbase = m0 + ((lane >= 16) ? 8 : 0);
  const int cx    = lane & 15;

#define EMIT_TILE(ACC, MT, NT)                                           \
  {                                                                      \
    int col = n0 + (NT) * 16 + cx;                                       \
    float bb = bias[col];                                                \
    _Pragma("unroll")                                                    \
    for (int r = 0; r < 8; ++r) {                                        \
      float v = ACC[r] + bb;                                             \
      if (RELU) v = fmaxf(v, 0.0f);                                      \
      size_t idx = (size_t)(rbase + (MT) * 16 + r) * (size_t)N           \
                 + (size_t)col;                                          \
      if (WF32)  Cf[idx] = v;                                            \
      if (WBF16) Cb[idx] = f2bf(v);                                      \
    }                                                                    \
  }
  EMIT_TILE(acc00, 0, 0)
  EMIT_TILE(acc01, 0, 1)
  EMIT_TILE(acc02, 0, 2)
  EMIT_TILE(acc03, 0, 3)
  EMIT_TILE(acc10, 1, 0)
  EMIT_TILE(acc11, 1, 1)
  EMIT_TILE(acc12, 1, 2)
  EMIT_TILE(acc13, 1, 3)
#undef EMIT_TILE
}

// ---------------------------------------------------------------------------
// LayerNorm over H=256; one block (256 thr) per row.
// MODE 0: x = X[row] + dummy[col]      -> write f32 + bf16
// MODE 1: x = X[row] + O[row]          -> LN, * mask[row], write f32 (d_out)
// ---------------------------------------------------------------------------
template <int MODE>
__global__ void ln_kernel(const float* __restrict__ X,
                          const float* __restrict__ R,       // dummy (MODE0) or o_f32 (MODE1)
                          const float* __restrict__ g,
                          const float* __restrict__ bvec,
                          const int*   __restrict__ mask,
                          float* __restrict__ outF,
                          unsigned short* __restrict__ outB) {
  int row = blockIdx.x;
  int c   = threadIdx.x;
  size_t idx = (size_t)row * H_ + c;
  float x = X[idx] + (MODE == 0 ? R[c] : R[idx]);

  __shared__ float red[H_];
  red[c] = x;
  __syncthreads();
  for (int s = 128; s > 0; s >>= 1) { if (c < s) red[c] += red[c + s]; __syncthreads(); }
  float mean = red[0] * (1.0f / H_);
  __syncthreads();
  float d = x - mean;
  red[c] = d * d;
  __syncthreads();
  for (int s = 128; s > 0; s >>= 1) { if (c < s) red[c] += red[c + s]; __syncthreads(); }
  float var = red[0] * (1.0f / H_);

  float y = d * rsqrtf(var + 1e-5f) * g[c] + bvec[c];
  if (MODE == 0) {
    outF[idx] = y;
    outB[idx] = f2bf(y);
  } else {
    outF[idx] = y * (float)mask[row];
  }
}

// ---------------------------------------------------------------------------
// host-side launcher
// ---------------------------------------------------------------------------
extern "C" void kernel_launch(void* const* d_in, const int* in_sizes, int n_in,
                              void* d_out, int out_size, void* d_ws, size_t ws_size,
                              hipStream_t stream) {
  (void)in_sizes; (void)n_in; (void)out_size; (void)ws_size;

  const float* tok        = (const float*)d_in[0];
  const int*   span_ids   = (const int*)d_in[1];
  const int*   span_masks = (const int*)d_in[2];
  /* d_in[3] = pooling (unused) */
  const float* dummy = (const float*)d_in[4];
  const float* Wq    = (const float*)d_in[5];
  const float* bq    = (const float*)d_in[6];
  const float* Wk    = (const float*)d_in[7];
  const float* bk    = (const float*)d_in[8];
  const float* Wv    = (const float*)d_in[9];
  const float* bv    = (const float*)d_in[10];
  const float* Wo    = (const float*)d_in[11];
  const float* bo    = (const float*)d_in[12];
  const float* ln_g  = (const float*)d_in[13];
  const float* ln_b  = (const float*)d_in[14];
  const float* w1    = (const float*)d_in[15];
  const float* b1    = (const float*)d_in[16];
  const float* w2    = (const float*)d_in[17];
  const float* b2    = (const float*)d_in[18];

  char* ws = (char*)d_ws;
  size_t off = 0;
  auto alloc = [&](size_t bytes) -> void* {
    void* p = ws + off;
    off = (off + bytes + 255) & ~(size_t)255;
    return p;
  };

  unsigned short* wk_bf  = (unsigned short*)alloc((size_t)H_ * H_ * 2);
  unsigned short* wv_bf  = (unsigned short*)alloc((size_t)H_ * H_ * 2);
  unsigned short* wo_bf  = (unsigned short*)alloc((size_t)H_ * H_ * 2);
  unsigned short* w1_bf  = (unsigned short*)alloc((size_t)FF_ * H_ * 2);
  unsigned short* w2_bf  = (unsigned short*)alloc((size_t)H_ * FF_ * 2);
  unsigned short* x_bf   = (unsigned short*)alloc((size_t)B_ * S_ * H_ * 2);
  float*          k_f    = (float*)alloc((size_t)B_ * S_ * H_ * 4);
  float*          v_f    = (float*)alloc((size_t)B_ * S_ * H_ * 4);
  float*          q_f    = (float*)alloc((size_t)H_ * 4);
  float*          sc_f   = (float*)alloc((size_t)B_ * NH_ * S_ * 4);
  unsigned short* ctx_bf = (unsigned short*)alloc((size_t)MROWS_ * H_ * 2);
  float*          attn_f = (float*)alloc((size_t)MROWS_ * H_ * 4);   // reused as ff
  float*          o_f    = (float*)alloc((size_t)MROWS_ * H_ * 4);
  unsigned short* o_bf   = (unsigned short*)alloc((size_t)MROWS_ * H_ * 2);
  unsigned short* h1_bf  = (unsigned short*)alloc((size_t)MROWS_ * FF_ * 2);

  // 1. weights -> bf16
  cvt_bf16_kernel<<<(H_ * H_ + 255) / 256, 256, 0, stream>>>(Wk, wk_bf, H_ * H_);
  cvt_bf16_kernel<<<(H_ * H_ + 255) / 256, 256, 0, stream>>>(Wv, wv_bf, H_ * H_);
  cvt_bf16_kernel<<<(H_ * H_ + 255) / 256, 256, 0, stream>>>(Wo, wo_bf, H_ * H_);
  cvt_bf16_kernel<<<(FF_ * H_ + 255) / 256, 256, 0, stream>>>(w1, w1_bf, FF_ * H_);
  cvt_bf16_kernel<<<(H_ * FF_ + 255) / 256, 256, 0, stream>>>(w2, w2_bf, H_ * FF_);

  // 2. x = tok + PE (bf16)
  prep_kernel<<<(B_ * S_ * H_) / 256, 256, 0, stream>>>(tok, x_bf);

  // 3. q projection (exact f32)
  qproj_kernel<<<1, 256, 0, stream>>>(dummy, Wq, bq, q_f);

  // 4/5. K and V projections: (2048 x 256) = (x 2048x256) * (W 256x256)^T
  gemm_wmma_kernel<false, true, false>
      <<<dim3((B_ * S_) / 256, H_ / 64), 256, 0, stream>>>(
          x_bf, wk_bf, bk, k_f, nullptr, B_ * S_, H_, H_);
  gemm_wmma_kernel<false, true, false>
      <<<dim3((B_ * S_) / 256, H_ / 64), 256, 0, stream>>>(
          x_bf, wv_bf, bv, v_f, nullptr, B_ * S_, H_, H_);

  // 6. attention scores
  scores_kernel<<<(B_ * NH_ * S_) / 256, 256, 0, stream>>>(q_f, k_f, sc_f);

  // 7. per-span softmax + weighted V  -> ctx (bf16)
  span_attn_kernel<<<MROWS_, 256, 0, stream>>>(sc_f, v_f, span_ids, ctx_bf);

  // 8. attn_out = ctx @ Wo^T + bo
  gemm_wmma_kernel<false, true, false>
      <<<dim3(MROWS_ / 256, H_ / 64), 256, 0, stream>>>(
          ctx_bf, wo_bf, bo, attn_f, nullptr, MROWS_, H_, H_);

  // 9. o = LN(attn_out + dummy)
  ln_kernel<0><<<MROWS_, 256, 0, stream>>>(attn_f, dummy, ln_g, ln_b, nullptr, o_f, o_bf);

  // 10. h1 = relu(o @ w1^T + b1)   (bf16 out)
  gemm_wmma_kernel<true, false, true>
      <<<dim3(MROWS_ / 256, FF_ / 64), 256, 0, stream>>>(
          o_bf, w1_bf, b1, nullptr, h1_bf, MROWS_, FF_, H_);

  // 11. ff = h1 @ w2^T + b2        (f32, reuse attn_f buffer)
  gemm_wmma_kernel<false, true, false>
      <<<dim3(MROWS_ / 256, H_ / 64), 256, 0, stream>>>(
          h1_bf, w2_bf, b2, attn_f, nullptr, MROWS_, H_, FF_);

  // 12. out = LN(ff + o) * mask
  ln_kernel<1><<<MROWS_, 256, 0, stream>>>(attn_f, o_f, ln_g, ln_b, span_masks,
                                           (float*)d_out, nullptr);
}
